// GeneralAttn_42064909697834
// MI455X (gfx1250) — matrix operand order repaired and checked by reading, not compiled, not run on hardware
//
#include <hip/hip_runtime.h>
#include <hip/hip_bf16.h>
#include <stdint.h>

// ---------------------------------------------------------------------------
// MI455X (gfx1250) fused multi-head attention, wave32 + WMMA bf16.
//   B=8 H=8 Np=1025 E=512 D=64.  All GEMMs via v_wmma_f32_16x16x32_bf16.
//   Roofline: attn_bias (269 MB, read once) ~11.5us @ 23.3 TB/s dominates;
//   everything else (~45 MB) is L2-resident (192 MB).  Bias is streamed with
//   double-buffered GLOBAL_LOAD_ASYNC_TO_LDS_B128 (ASYNCcnt) + NT hint so it
//   overlaps the WMMAs and does not evict the resident working set.
// ---------------------------------------------------------------------------

typedef __attribute__((ext_vector_type(16))) __bf16 v16bf;
typedef __attribute__((ext_vector_type(8)))  float  v8f;

#define NB   8
#define NH   8
#define NPX  1025          // sequence length incl. graph token
#define EMB  512
#define HD   64
#define NPAD 1056          // keys padded to 33*32
#define NPQ  1040          // query rows padded to 65*16
#define KSTEPS (NPAD / 32) // 33

__device__ __forceinline__ int imin(int a, int b) { return a < b ? a : b; }

__device__ __forceinline__ v8f wmma_bf16(v16bf a, v16bf b, v8f c) {
  // (neg_a, A, neg_b, B, c_mod, C, reuse_a, reuse_b)
  return __builtin_amdgcn_wmma_f32_16x16x32_bf16(false, a, false, b, (short)0, c,
                                                 false, false);
}

// A-fragment (16x32, 16-bit): lane half g holds two 8-element K chunks of row
// (lane&15): K = g*8..g*8+7 and K = 16+g*8..16+g*8+7  (ISA 7.12.2).
__device__ __forceinline__ v16bf load_a(const __bf16* base, int ld, int row,
                                        int kb, int g) {
  const __bf16* r = base + (size_t)row * ld + kb + g * 8;
  union { uint4 q[2]; v16bf v; } u;
  u.q[0] = *(const uint4*)(r);
  u.q[1] = *(const uint4*)(r + 16);
  return u.v;
}

// B-fragment (32x16, 16-bit): lane (n = lane&15) holds column n; lane half g
// holds K = g*8..g*8+7 and K = 16+g*8..16+g*8+7.  colbase points at column n, K=0.
__device__ __forceinline__ v16bf load_b(const __bf16* colbase, int g) {
  const __bf16* r = colbase + g * 8;
  union { uint4 q[2]; v16bf v; } u;
  u.q[0] = *(const uint4*)(r);
  u.q[1] = *(const uint4*)(r + 16);
  return u.v;
}

// A-fragment built on the fly from fp32 row-major data (RNE via cast).
__device__ __forceinline__ v16bf cvt_a_f32(const float* rowp, int kb, int g) {
  const float* c0 = rowp + kb + g * 8;
  const float* c1 = rowp + kb + 16 + g * 8;
  float4 a0 = *(const float4*)(c0);
  float4 a1 = *(const float4*)(c0 + 4);
  float4 b0 = *(const float4*)(c1);
  float4 b1 = *(const float4*)(c1 + 4);
  v16bf v;
  v[0] = (__bf16)a0.x; v[1] = (__bf16)a0.y; v[2]  = (__bf16)a0.z; v[3]  = (__bf16)a0.w;
  v[4] = (__bf16)a1.x; v[5] = (__bf16)a1.y; v[6]  = (__bf16)a1.z; v[7]  = (__bf16)a1.w;
  v[8] = (__bf16)b0.x; v[9] = (__bf16)b0.y; v[10] = (__bf16)b0.z; v[11] = (__bf16)b0.w;
  v[12] = (__bf16)b1.x; v[13] = (__bf16)b1.y; v[14] = (__bf16)b1.z; v[15] = (__bf16)b1.w;
  return v;
}

// Async-copy one 16x32 fp32 bias tile (2 KB) to LDS: 4 x ASYNC_TO_LDS_B128,
// 32 lanes x 16 B each.  Lane covers (row = i*4 + lane/8, 4 cols at (lane%8)*4).
// Rows/cols clamped into the [1025,1025] bias (clamped slots are masked -inf
// later; boundary chunk at col 1024 over-reads <=12 B inside the allocation).
__device__ __forceinline__ void async_bias_tile(const float* biasb, int q0,
                                                int k0, float* dst, int lane) {
  const int rsub = lane >> 3;          // 0..3
  const int csub = (lane & 7) << 2;    // 0,4,..,28
  const int colc = imin(k0 + csub, NPX - 1);
#pragma unroll
  for (int i = 0; i < 4; i++) {
    const int qc = imin(q0 + i * 4 + rsub, NPX - 1);
    const float* g = biasb + (size_t)qc * NPX + colc;
    // generic LDS pointer: low 32 bits are the LDS byte offset (ISA 10.2)
    unsigned l = (unsigned)(uintptr_t)(dst + (i * 4 + rsub) * 32 + csub);
    asm volatile("global_load_async_to_lds_b128 %0, %1, off th:TH_LOAD_NT"
                 :: "v"(l), "v"(g) : "memory");
  }
}

__device__ __forceinline__ void wait_async_le4() {
  asm volatile("s_wait_asynccnt 0x4" ::: "memory");
}
__device__ __forceinline__ void wait_async_0() {
  asm volatile("s_wait_asynccnt 0x0" ::: "memory");
}

// ---------------------------------------------------------------------------
// Kernel 0a: convert the 4 weight matrices (512x512 fp32) to bf16 once.
// ---------------------------------------------------------------------------
__global__ void convert_w_kernel(const float* __restrict__ wq,
                                 const float* __restrict__ wk,
                                 const float* __restrict__ wv,
                                 const float* __restrict__ wo,
                                 __bf16* __restrict__ out) {
  int i = blockIdx.x * blockDim.x + threadIdx.x;   // 0 .. 4*262144-1
  int m = i >> 18;
  int e = i & 262143;
  const float* src = (m == 0) ? wq : (m == 1) ? wk : (m == 2) ? wv : wo;
  out[((size_t)m << 18) + e] = (__bf16)src[e];
}

// ---------------------------------------------------------------------------
// Kernel 0b: pack the padded keep-mask into bits: mb[b][q][w] bit j = keep for
// key k = 32w + j (k=0 always kept; 1<=k<=1024 from pad_mask[q-1][k-1];
// k>=1025 dropped; q=0 row all-kept).  1.1 MB total -> L2 resident.
// ---------------------------------------------------------------------------
__global__ void maskbits_kernel(const unsigned char* __restrict__ pmask,
                                unsigned int* __restrict__ mb) {
  int i = blockIdx.x * blockDim.x + threadIdx.x;
  if (i >= NB * NPQ * KSTEPS) return;
  int w = i % KSTEPS;
  int row = i / KSTEPS;
  int q = row % NPQ;
  int b = row / NPQ;
  unsigned int bits = 0;
  for (int j = 0; j < 32; j++) {
    int k = w * 32 + j;
    int keep;
    if (k == 0) keep = 1;
    else if (k >= NPX) keep = 0;
    else if (q == 0 || q >= NPX) keep = 1;   // graph-token row / pad rows
    else keep = pmask[((size_t)b * 1024 + (q - 1)) * 1024 + (k - 1)] ? 1 : 0;
    bits |= (unsigned int)keep << j;
  }
  mb[i] = bits;
}

// ---------------------------------------------------------------------------
// Kernel 1: QKV projection.  One wave per (16-row, 16-col) tile, reusing the
// x A-fragment across the three weight matrices (x and the bf16 weights are
// L2-resident, so the re-reads never hit HBM).  Q pre-scaled by 1/8 (exact in
// bf16).  K row-major [bh,np,d]; V transposed [bh,d,np] for the PV B-operand.
// ---------------------------------------------------------------------------
__global__ void __launch_bounds__(32)
qkv_kernel(const float* __restrict__ x,
           const __bf16* __restrict__ wqb, const __bf16* __restrict__ wkb,
           const __bf16* __restrict__ wvb,
           const float* __restrict__ bq, const float* __restrict__ bk,
           const float* __restrict__ bv,
           __bf16* __restrict__ Qm, __bf16* __restrict__ Km,
           __bf16* __restrict__ VT) {
  const int lane = threadIdx.x & 31;
  const int g = lane >> 4, mr = lane & 15;
  const int ct = blockIdx.x & 31;                  // 32 column tiles of E
  const int rt = blockIdx.x >> 5;                  // 0 .. 8*66-1 row tiles
  const int b = rt / (NPAD / 16);
  const int np0 = (rt % (NPAD / 16)) * 16;
  const int xrow = b * NPX + imin(np0 + mr, NPX - 1);  // clamp pad rows
  const float* xr = x + (size_t)xrow * EMB;
  const int c0 = ct * 16;
  const __bf16* wq_c = wqb + (size_t)(c0 + mr) * EMB;  // col of W.T = row of W
  const __bf16* wk_c = wkb + (size_t)(c0 + mr) * EMB;
  const __bf16* wv_c = wvb + (size_t)(c0 + mr) * EMB;

  v8f aq = {}, ak = {}, av = {};
  for (int kk = 0; kk < EMB; kk += 32) {
    v16bf a = cvt_a_f32(xr, kk, g);
    aq = wmma_bf16(a, load_b(wq_c + kk, g), aq);
    ak = wmma_bf16(a, load_b(wk_c + kk, g), ak);
    av = wmma_bf16(a, load_b(wv_c + kk, g), av);
  }
  for (int r = 0; r < 8; r++) {
    const int np = np0 + r + g * 8;      // C layout: row = r + 8*(lane/16)
    const int col = c0 + mr;             //           col = lane & 15
    const int h = col >> 6, d = col & 63;
    const size_t bh = (size_t)(b * NH + h);
    float qv = (aq[r] + bq[col]) * 0.125f;   // 1/sqrt(64) folded in
    float kv = ak[r] + bk[col];
    float vv = av[r] + bv[col];
    Qm[(bh * NPAD + np) * HD + d] = (__bf16)qv;
    Km[(bh * NPAD + np) * HD + d] = (__bf16)kv;
    VT[(bh * HD + d) * NPAD + np] = (__bf16)vv;
  }
}

// ---------------------------------------------------------------------------
// Kernel 2: flash attention.  One wave per (b, h, 16-query tile); 33 k-steps
// of 32 keys.  Scores live only in VGPRs; the bias tile for step ks+1 streams
// HBM->LDS via async-to-LDS while step ks runs its 8 WMMAs; P is transposed
// C->A layout through 1 KB of LDS.
// ---------------------------------------------------------------------------
__global__ void __launch_bounds__(32)
attn_kernel(const __bf16* __restrict__ Qm, const __bf16* __restrict__ Km,
            const __bf16* __restrict__ VT, const float* __restrict__ bias,
            const unsigned int* __restrict__ mb, __bf16* __restrict__ O) {
  __shared__ float btile[2][16 * 32];   // bias tile double buffer (2 x 2 KB)
  __shared__ __bf16 plds[16 * 32];      // P transpose bounce (1 KB)
  const int lane = threadIdx.x & 31;
  const int g = lane >> 4, mr = lane & 15;
  const int qt = blockIdx.x % 65;
  const int bh = blockIdx.x / 65;                  // 0..63
  const int h = bh & 7, b = bh >> 3;
  const int q0 = qt * 16;
  const __bf16* Qb = Qm + (size_t)bh * NPAD * HD;
  const __bf16* Kb = Km + (size_t)bh * NPAD * HD;
  const __bf16* Vb = VT + (size_t)bh * HD * NPAD;
  const float* biasb = bias + (size_t)bh * NPX * NPX;

  const v16bf qf0 = load_a(Qb, HD, q0 + mr, 0, g);   // d = 0..31
  const v16bf qf1 = load_a(Qb, HD, q0 + mr, 32, g);  // d = 32..63

  // per-row keep-bit words: word ks holds bits for keys 32ks..32ks+31
  const unsigned int* mrow[8];
#pragma unroll
  for (int r = 0; r < 8; r++)
    mrow[r] = mb + ((size_t)b * NPQ + (q0 + r + g * 8)) * KSTEPS;

  v8f acc0 = {}, acc1 = {}, acc2 = {}, acc3 = {};
  float M[8], L[8];
#pragma unroll
  for (int r = 0; r < 8; r++) { M[r] = -__builtin_inff(); L[r] = 0.f; }

  async_bias_tile(biasb, q0, 0, &btile[0][0], lane);   // prefetch tile 0

  for (int ks = 0; ks < KSTEPS; ks++) {
    const int k0 = ks * 32;
    if (ks + 1 < KSTEPS)                 // prefetch next tile into other buffer
      async_bias_tile(biasb, q0, k0 + 32, &btile[(ks + 1) & 1][0], lane);

    // ---- S = (Q/8) K^T for 32 keys: two 16x16 tiles ----
    v8f s0 = {}, s1 = {};
    {
      const __bf16* kc0 = Kb + (size_t)(k0 + mr) * HD;        // key row k0+n
      const __bf16* kc1 = Kb + (size_t)(k0 + 16 + mr) * HD;
      s0 = wmma_bf16(qf0, load_b(kc0, g), s0);
      s0 = wmma_bf16(qf1, load_b(kc0 + 32, g), s0);
      s1 = wmma_bf16(qf0, load_b(kc1, g), s1);
      s1 = wmma_bf16(qf1, load_b(kc1 + 32, g), s1);
    }

    // async loads complete in order: <=4 outstanding => tile ks is in LDS
    if (ks + 1 < KSTEPS) wait_async_le4(); else wait_async_0();
    const float* bt = &btile[ks & 1][0];

    // ---- bias + pad mask + online softmax ----
#pragma unroll
    for (int r = 0; r < 8; r++) {
      const int m = r + g * 8;
      const unsigned int wbits = mrow[r][ks];
      float v0 = s0[r] + bt[m * 32 + mr] +
                 (((wbits >> mr) & 1u) ? 0.f : -__builtin_inff());
      float v1 = s1[r] + bt[m * 32 + 16 + mr] +
                 (((wbits >> (16 + mr)) & 1u) ? 0.f : -__builtin_inff());
      // row max over the 16 lanes of this half (xor 1,2,4,8 stays in-half)
      float t = fmaxf(v0, v1);
      t = fmaxf(t, __shfl_xor(t, 1));
      t = fmaxf(t, __shfl_xor(t, 2));
      t = fmaxf(t, __shfl_xor(t, 4));
      t = fmaxf(t, __shfl_xor(t, 8));
      float mnew = fmaxf(M[r], t);
      float alpha = __expf(M[r] - mnew);   // k=0 never masked -> mnew finite
      float p0 = __expf(v0 - mnew);
      float p1 = __expf(v1 - mnew);
      float rs = p0 + p1;
      rs += __shfl_xor(rs, 1);
      rs += __shfl_xor(rs, 2);
      rs += __shfl_xor(rs, 4);
      rs += __shfl_xor(rs, 8);
      L[r] = L[r] * alpha + rs;
      M[r] = mnew;
      acc0[r] *= alpha; acc1[r] *= alpha; acc2[r] *= alpha; acc3[r] *= alpha;
      plds[m * 32 + mr] = (__bf16)p0;          // C layout -> row-major 16x32
      plds[m * 32 + 16 + mr] = (__bf16)p1;
    }
    __syncthreads();
    // ---- O += P V  (K-dim = 32 keys; 4 output d-tiles) ----
    {
      const v16bf pf = load_a(plds, 32, mr, 0, g);
      const __bf16* vc = Vb + (size_t)mr * NPAD + k0;   // column d = d0 + n
      acc0 = wmma_bf16(pf, load_b(vc, g), acc0);
      acc1 = wmma_bf16(pf, load_b(vc + 16 * NPAD, g), acc1);
      acc2 = wmma_bf16(pf, load_b(vc + 32 * NPAD, g), acc2);
      acc3 = wmma_bf16(pf, load_b(vc + 48 * NPAD, g), acc3);
    }
    __syncthreads();
  }
  // ---- normalize and store O (bf16, row-major [b, q, h*64+d]) ----
#pragma unroll
  for (int r = 0; r < 8; r++) {
    const int q = q0 + r + g * 8;
    if (q < NPX) {
      const float inv = 1.0f / L[r];
      const size_t o = ((size_t)b * NPQ + q) * EMB + h * HD;
      O[o + mr]      = (__bf16)(acc0[r] * inv);
      O[o + 16 + mr] = (__bf16)(acc1[r] * inv);
      O[o + 32 + mr] = (__bf16)(acc2[r] * inv);
      O[o + 48 + mr] = (__bf16)(acc3[r] * inv);
    }
  }
}

// ---------------------------------------------------------------------------
// Kernel 3: output projection  out = O @ Wo.T + bo  (fp32 result).
// ---------------------------------------------------------------------------
__global__ void __launch_bounds__(32)
outproj_kernel(const __bf16* __restrict__ O, const __bf16* __restrict__ wob,
               const float* __restrict__ bo, float* __restrict__ out) {
  const int lane = threadIdx.x & 31;
  const int g = lane >> 4, mr = lane & 15;
  const int ct = blockIdx.x & 31;
  const int rt = blockIdx.x >> 5;                  // 0 .. 8*65-1
  const int b = rt / 65;
  const int q0 = (rt % 65) * 16;
  const int arow = b * NPQ + q0 + mr;
  const int c0 = ct * 16;
  const __bf16* wo_c = wob + (size_t)(c0 + mr) * EMB;
  v8f acc = {};
  for (int kk = 0; kk < EMB; kk += 32) {
    v16bf a = load_a(O, EMB, arow, kk, g);
    acc = wmma_bf16(a, load_b(wo_c + kk, g), acc);
  }
  for (int r = 0; r < 8; r++) {
    const int q = q0 + r + g * 8;
    if (q < NPX) {
      const int col = c0 + mr;
      out[((size_t)b * NPX + q) * EMB + col] = acc[r] + bo[col];
    }
  }
}

// ---------------------------------------------------------------------------
// Launch.  Inputs: 0:x 1:attn_bias 2:pad_mask 3:Wq 4:bq 5:Wk 6:bk 7:Wv 8:bv
// 9:Wo 10:bo.   Workspace (~38 MB): 4 bf16 weights | Q | K | V^T | O | maskbits
// ---------------------------------------------------------------------------
extern "C" void kernel_launch(void* const* d_in, const int* in_sizes, int n_in,
                              void* d_out, int out_size, void* d_ws,
                              size_t ws_size, hipStream_t stream) {
  const float* x    = (const float*)d_in[0];
  const float* bias = (const float*)d_in[1];
  const unsigned char* pmask = (const unsigned char*)d_in[2];  // jnp.bool_
  const float* Wq = (const float*)d_in[3];
  const float* bq = (const float*)d_in[4];
  const float* Wk = (const float*)d_in[5];
  const float* bk = (const float*)d_in[6];
  const float* Wv = (const float*)d_in[7];
  const float* bv = (const float*)d_in[8];
  const float* Wo = (const float*)d_in[9];
  const float* bo = (const float*)d_in[10];
  float* out = (float*)d_out;

  char* ws = (char*)d_ws;
  const size_t W_BYTES  = (size_t)EMB * EMB * 2;                 // 512 KB
  const size_t QK_BYTES = (size_t)NB * NH * NPAD * HD * 2;       // 8.65 MB
  const size_t O_BYTES  = (size_t)NB * NPQ * EMB * 2;            // 8.52 MB
  __bf16* Wb  = (__bf16*)ws;                                     // [4][512*512]
  __bf16* WQb = Wb;
  __bf16* WKb = (__bf16*)(ws + 1 * W_BYTES);
  __bf16* WVb = (__bf16*)(ws + 2 * W_BYTES);
  __bf16* WOb = (__bf16*)(ws + 3 * W_BYTES);
  __bf16* Qm  = (__bf16*)(ws + 4 * W_BYTES);
  __bf16* Km  = (__bf16*)(ws + 4 * W_BYTES + QK_BYTES);
  __bf16* VT  = (__bf16*)(ws + 4 * W_BYTES + 2 * QK_BYTES);
  __bf16* Om  = (__bf16*)(ws + 4 * W_BYTES + 3 * QK_BYTES);
  unsigned int* MB =
      (unsigned int*)(ws + 4 * W_BYTES + 3 * QK_BYTES + O_BYTES);

  // 0a: weights fp32 -> bf16 (4 * 512*512 elements)
  convert_w_kernel<<<4096, 256, 0, stream>>>(Wq, Wk, Wv, Wo, Wb);
  // 0b: keep-mask bit words (8*1040*33)
  maskbits_kernel<<<(NB * NPQ * KSTEPS + 255) / 256, 256, 0, stream>>>(pmask, MB);
  // 1: QKV projection: (8*66 row tiles) * (32 col tiles), one wave each
  qkv_kernel<<<(NB * (NPAD / 16)) * 32, 32, 0, stream>>>(
      x, WQb, WKb, WVb, bq, bk, bv, Qm, Km, VT);
  // 2: flash attention: 64 (b,h) * 65 q-tiles
  attn_kernel<<<NB * NH * 65, 32, 0, stream>>>(Qm, Km, VT, bias, MB, Om);
  // 3: output projection: (8*65 row tiles) * (32 col tiles)
  outproj_kernel<<<(NB * 65) * 32, 32, 0, stream>>>(Om, WOb, bo, out);
}
